// Tacotron2_34437047779866
// MI455X (gfx1250) — matrix operand order repaired and checked by reading, not compiled, not run on hardware
//
#include <hip/hip_runtime.h>
#include <hip/hip_bf16.h>

typedef __attribute__((ext_vector_type(16))) _Float16 v16h;
typedef __attribute__((ext_vector_type(8)))  _Float16 v8h;
typedef __attribute__((ext_vector_type(8)))  float    v8f;

// Model dims
constexpr int cB = 32, cS = 256, cT = 500;
constexpr int cEMB = 512, cENC = 512, cH = 256, cDEC = 1024, cATT = 128, cNMEL = 80;
constexpr int cL = 3 * cT;            // 1500 output frames
constexpr int cK1 = cEMB * 5;         // 2560 encoder conv im2col K
constexpr int cKP = 416;              // postnet conv1 K (80*5=400 padded to mult of 32)

__device__ __forceinline__ float sigm(float x) { return 1.0f / (1.0f + __expf(-x)); }

// ---------------------------------------------------------------------------
// WMMA operand loaders (contiguous k order, same for A and B):
//  A 16x32 f16: lane<16 -> row M=lane, 16 contiguous K at 0..15;
//               lane>=16 -> row M=lane-16, 16 contiguous K at 16..31.
//  B 32x16 f16: lane holds column N=lane&15, 16 contiguous K at (lane>>4)*16,
//               i.e. a contiguous 32B load from the (N,K) row-major weight row.
//  C/D 16x16 f32: VGPR r holds rows {r, r+8}; lane&15 = column.
// ---------------------------------------------------------------------------

// Generic WMMA GEMM, 1 wave = one 16x16 tile.  C = act(A @ W^T (+bias) (+=C))
template <int ACT, int HASBIAS, int ACCUM>
__global__ void k_gemm(const _Float16* __restrict__ A, const _Float16* __restrict__ W,
                       const float* __restrict__ bias, float* __restrict__ C,
                       int M, int N, int K) {
  const int lane = threadIdx.x;                 // blockDim = 32 (one wave)
  const int tn = blockIdx.x, tm = blockIdx.y;
  const int arow = lane & 15;
  const int ah   = (lane >> 4) & 1;
  const _Float16* Ar = A + (size_t)(tm * 16 + arow) * K + ah * 16;
  const _Float16* Wr = W + (size_t)(tn * 16 + (lane & 15)) * K + ah * 16;
  v8f acc = {};
  for (int k = 0; k < K; k += 32) {
    v16h a, bm;
    ((v8h*)&a)[0] = *(const v8h*)(Ar + k);
    ((v8h*)&a)[1] = *(const v8h*)(Ar + k + 8);
    bm = *(const v16h*)(Wr + k);
    if (k + 32 < K) {
      __builtin_prefetch(Ar + k + 32, 0, 1);    // -> global_prefetch_b8
      __builtin_prefetch(Wr + k + 32, 0, 1);
    }
    acc = __builtin_amdgcn_wmma_f32_16x16x32_f16(false, a, false, bm, (short)0, acc,
                                                 false, false);
  }
  const int nloc = lane & 15;
  const int mofs = (lane >> 4) << 3;
#pragma unroll
  for (int r = 0; r < 8; ++r) {
    int m = tm * 16 + r + mofs;
    int n = tn * 16 + nloc;
    float v = acc[r];
    if (HASBIAS) v += bias[n];
    if (ACCUM) v += C[(size_t)m * N + n];
    if (ACT == 1) v = fmaxf(v, 0.0f);
    C[(size_t)m * N + n] = v;
  }
}

// 1x4 N-tile WMMA GEMM: one wave computes a 16x64 strip; the A chunk loaded
// once per K step feeds 4 WMMAs.  grid.x = N/64, grid.y = M/16.  N % 64 == 0.
template <int ACT, int HASBIAS, int ACCUM>
__global__ void k_gemm4(const _Float16* __restrict__ A, const _Float16* __restrict__ W,
                        const float* __restrict__ bias, float* __restrict__ C,
                        int M, int N, int K) {
  const int lane = threadIdx.x;                 // blockDim = 32
  const int tn0 = blockIdx.x * 4, tm = blockIdx.y;
  const int arow = lane & 15;
  const int ah   = (lane >> 4) & 1;
  const _Float16* Ar  = A + (size_t)(tm * 16 + arow) * K + ah * 16;
  const _Float16* Wr0 = W + (size_t)((tn0 + 0) * 16 + (lane & 15)) * K + ah * 16;
  const _Float16* Wr1 = Wr0 + (size_t)16 * K;
  const _Float16* Wr2 = Wr0 + (size_t)32 * K;
  const _Float16* Wr3 = Wr0 + (size_t)48 * K;
  v8f acc[4] = {};
#pragma unroll 1
  for (int k = 0; k < K; k += 32) {
    v16h a;
    ((v8h*)&a)[0] = *(const v8h*)(Ar + k);
    ((v8h*)&a)[1] = *(const v8h*)(Ar + k + 8);
    v16h b0 = *(const v16h*)(Wr0 + k);
    v16h b1 = *(const v16h*)(Wr1 + k);
    v16h b2 = *(const v16h*)(Wr2 + k);
    v16h b3 = *(const v16h*)(Wr3 + k);
    if (k + 32 < K) {
      __builtin_prefetch(Ar + k + 32, 0, 1);
      __builtin_prefetch(Wr0 + k + 32, 0, 1);
      __builtin_prefetch(Wr3 + k + 32, 0, 1);
    }
    acc[0] = __builtin_amdgcn_wmma_f32_16x16x32_f16(false, a, false, b0, (short)0, acc[0], false, false);
    acc[1] = __builtin_amdgcn_wmma_f32_16x16x32_f16(false, a, false, b1, (short)0, acc[1], false, false);
    acc[2] = __builtin_amdgcn_wmma_f32_16x16x32_f16(false, a, false, b2, (short)0, acc[2], false, false);
    acc[3] = __builtin_amdgcn_wmma_f32_16x16x32_f16(false, a, false, b3, (short)0, acc[3], false, false);
  }
  const int nloc = lane & 15;
  const int mofs = (lane >> 4) << 3;
#pragma unroll
  for (int j = 0; j < 4; ++j) {
#pragma unroll
    for (int r = 0; r < 8; ++r) {
      int m = tm * 16 + r + mofs;
      int n = (tn0 + j) * 16 + nloc;
      float v = acc[j][r];
      if (HASBIAS) v += bias[n];
      if (ACCUM) v += C[(size_t)m * N + n];
      if (ACT == 1) v = fmaxf(v, 0.0f);
      C[(size_t)m * N + n] = v;
    }
  }
}

// ---------------------------------------------------------------------------
// Persistent bi-LSTM recurrence: 1 workgroup per direction (blockIdx.x = dir),
// 32 waves, h (f16) / c (f32) / gate pre-activations (f32) in LDS (176KB).
// A wave's 4 tiles t = wv+32*i share tm = wv&1, so each step does:
//   K-loop { 1 LDS A-load -> 4 WMMAs } with B rolled (#pragma unroll 1) so the
//   compiler cannot hoist the step-invariant Whh tiles into (spilled) regs.
// ---------------------------------------------------------------------------
__global__ void __launch_bounds__(1024)
k_lstm(const float* __restrict__ xwF, const float* __restrict__ xwB,
       const _Float16* __restrict__ whhF, const _Float16* __restrict__ whhB,
       _Float16* __restrict__ xs_next, float* __restrict__ hs32) {
  extern __shared__ char smem[];
  _Float16* h16 = (_Float16*)smem;                                   // 32*256 f16
  float* cbuf = (float*)(smem + cB * cH * sizeof(_Float16));         // 32*256 f32
  float* gbuf = (float*)(smem + cB * cH * (sizeof(_Float16) + sizeof(float))); // 32*1024

  const int dir = blockIdx.x;
  const float* xw0 = dir ? xwB : xwF;
  const _Float16* whh = dir ? whhB : whhF;
  const int tid = threadIdx.x;
  for (int p = tid; p < cB * cH; p += 1024) { h16[p] = (_Float16)0.0f; cbuf[p] = 0.0f; }

  const int lane = tid & 31, wv = tid >> 5;
  const int arow = lane & 15, ah = (lane >> 4) & 1;
  const int nloc = lane & 15, mofs = (lane >> 4) << 3;
  const int tm  = wv & 1;          // shared M-tile for this wave's 4 tiles
  const int tnb = wv >> 1;         // N tiles: tnb + {0,16,32,48}
  const _Float16* hr  = h16 + (tm * 16 + arow) * cH + ah * 16;
  const _Float16* wr0 = whh + (size_t)((tnb +  0) * 16 + nloc) * cH + ah * 16;
  const _Float16* wr1 = whh + (size_t)((tnb + 16) * 16 + nloc) * cH + ah * 16;
  const _Float16* wr2 = whh + (size_t)((tnb + 32) * 16 + nloc) * cH + ah * 16;
  const _Float16* wr3 = whh + (size_t)((tnb + 48) * 16 + nloc) * cH + ah * 16;

  for (int step = 0; step < cS; ++step) {
    const int s = dir ? (cS - 1 - step) : step;
    __syncthreads();                              // prev-step h16/cbuf visible
    const float* xw = xw0 + (size_t)s * cB * (4 * cH);
    v8f acc[4] = {};
#pragma unroll 1
    for (int k = 0; k < cH; k += 32) {
      v16h a;
      ((v8h*)&a)[0] = *(const v8h*)(hr + k);
      ((v8h*)&a)[1] = *(const v8h*)(hr + k + 8);
      v16h b0 = *(const v16h*)(wr0 + k);
      v16h b1 = *(const v16h*)(wr1 + k);
      v16h b2 = *(const v16h*)(wr2 + k);
      v16h b3 = *(const v16h*)(wr3 + k);
      acc[0] = __builtin_amdgcn_wmma_f32_16x16x32_f16(false, a, false, b0, (short)0, acc[0], false, false);
      acc[1] = __builtin_amdgcn_wmma_f32_16x16x32_f16(false, a, false, b1, (short)0, acc[1], false, false);
      acc[2] = __builtin_amdgcn_wmma_f32_16x16x32_f16(false, a, false, b2, (short)0, acc[2], false, false);
      acc[3] = __builtin_amdgcn_wmma_f32_16x16x32_f16(false, a, false, b3, (short)0, acc[3], false, false);
    }
#pragma unroll
    for (int j = 0; j < 4; ++j) {
      const int tn = tnb + 16 * j;
#pragma unroll
      for (int r = 0; r < 8; ++r) {
        int m = tm * 16 + r + mofs;               // batch index 0..31
        int n = tn * 16 + nloc;                   // gate index 0..1023
        gbuf[m * (4 * cH) + n] = acc[j][r] + xw[m * (4 * cH) + n];
      }
    }
    __syncthreads();
#pragma unroll
    for (int e = 0; e < 8; ++e) {                 // 8192 cell states / 1024 thr
      int p = e * 1024 + tid;
      int b = p >> 8, j = p & 255;
      const float* g = gbuf + b * (4 * cH);
      float c = sigm(g[cH + j]) * cbuf[p] + sigm(g[j]) * tanhf(g[2 * cH + j]);
      float h = sigm(g[3 * cH + j]) * tanhf(c);
      cbuf[p] = c;
      h16[b * cH + j] = (_Float16)h;
      size_t oi = ((size_t)(s * cB + b)) * cENC + dir * cH + j;
      xs_next[oi] = (_Float16)h;
      hs32[oi] = h;
    }
  }
}

// ---------------------------------------------------------------------------
// small helpers
// ---------------------------------------------------------------------------
__global__ void k_cvt_f16(const float* __restrict__ s, _Float16* __restrict__ d, int n) {
  int i = blockIdx.x * 256 + threadIdx.x;
  if (i < n) d[i] = (_Float16)s[i];
}
__global__ void k_cvt_pad(const float* __restrict__ s, _Float16* __restrict__ d,
                          int rows, int incols, int outcols) {
  int i = blockIdx.x * 256 + threadIdx.x;
  if (i >= rows * outcols) return;
  int r = i / outcols, c = i % outcols;
  d[i] = (c < incols) ? (_Float16)s[r * incols + c] : (_Float16)0.0f;
}
__global__ void k_add_vec(const float* a, const float* b, float* o, int n) {
  int i = blockIdx.x * 256 + threadIdx.x;
  if (i < n) o[i] = a[i] + b[i];
}
__global__ void k_zero_f32(float* p, int n) {
  int i = blockIdx.x * 256 + threadIdx.x;
  if (i < n) p[i] = 0.0f;
}
__global__ void k_zero_f16(_Float16* p, int n) {
  int i = blockIdx.x * 256 + threadIdx.x;
  if (i < n) p[i] = (_Float16)0.0f;
}

// encoder: embedding gather fused with conv1 im2col; rows (b,s), cols i*5+k
__global__ void k_emb_im2col(const int* __restrict__ text, const float* __restrict__ emb,
                             _Float16* __restrict__ X) {
  size_t idx = (size_t)blockIdx.x * 256 + threadIdx.x;
  if (idx >= (size_t)cB * cS * cK1) return;
  int row = (int)(idx / cK1), col = (int)(idx % cK1);
  int b = row >> 8, s = row & 255;
  int i = col / 5, k = col % 5;
  int ss = s + k - 2;
  float v = 0.0f;
  if (ss >= 0 && ss < cS) v = emb[(size_t)text[b * cS + ss] * cEMB + i];
  X[idx] = (_Float16)v;
}

// batchnorm over rows for 512 channels (columns); training-mode stats
__global__ void k_bn_stats(const float* __restrict__ X, int rows,
                           float* __restrict__ s1, float* __restrict__ s2) {
  __shared__ float r1[256], r2[256];
  int c = blockIdx.x, tid = threadIdx.x;
  float a = 0.0f, b = 0.0f;
  for (int r = tid; r < rows; r += 256) {
    float x = X[(size_t)r * 512 + c];
    a += x; b += x * x;
  }
  r1[tid] = a; r2[tid] = b; __syncthreads();
  for (int off = 128; off > 0; off >>= 1) {
    if (tid < off) { r1[tid] += r1[tid + off]; r2[tid] += r2[tid + off]; }
    __syncthreads();
  }
  if (tid == 0) { s1[c] = r1[0]; s2[c] = r2[0]; }
}
__global__ void k_bn_apply(float* __restrict__ X, int rows,
                           const float* __restrict__ s1, const float* __restrict__ s2,
                           const float* __restrict__ g, const float* __restrict__ bb,
                           int doTanh) {
  size_t idx = (size_t)blockIdx.x * 256 + threadIdx.x;
  if (idx >= (size_t)rows * 512) return;
  int c = (int)(idx & 511);
  float m = s1[c] / rows;
  float v = s2[c] / rows - m * m;
  float y = (X[idx] - m) * rsqrtf(v + 1e-5f) * g[c] + bb[c];
  X[idx] = doTanh ? tanhf(y) : y;
}

// (b*S+s, c) f32  ->  (s*B+b, c) f16
__global__ void k_enc_transpose(const float* __restrict__ c1, _Float16* __restrict__ xs) {
  size_t idx = (size_t)blockIdx.x * 256 + threadIdx.x;
  if (idx >= (size_t)cB * cS * cENC) return;
  int row = (int)(idx / cENC), c = (int)(idx % cENC);
  int b = row >> 8, s = row & 255;
  xs[((size_t)(s * cB + b)) * cENC + c] = (_Float16)c1[idx];
}

// decoder prenet (per-batch block): din -> relu(W1)->relu(W2) -> ri_h[:, 0:256]
__global__ void k_prenet(const float* __restrict__ mels, int t,
                         const float* __restrict__ w1, const float* __restrict__ b1,
                         const float* __restrict__ w2, const float* __restrict__ b2,
                         _Float16* __restrict__ ri_h) {
  __shared__ float din[cNMEL];
  __shared__ float t1[256];
  int b = blockIdx.x, tid = threadIdx.x;
  if (tid < cNMEL)
    din[tid] = (t == 0) ? 0.0f : mels[(size_t)b * cNMEL * cT + tid * cT + (t - 1)];
  __syncthreads();
  float a = b1[tid];
  for (int k = 0; k < cNMEL; ++k) a += w1[tid * cNMEL + k] * din[k];
  t1[tid] = fmaxf(a, 0.0f);
  __syncthreads();
  float a2 = b2[tid];
  for (int k = 0; k < 256; ++k) a2 += w2[tid * 256 + k] * t1[k];
  ri_h[b * 768 + tid] = (_Float16)fmaxf(a2, 0.0f);
}

// Bahdanau attention, one block per batch row; ctx -> ri_h[:,256:768], hc_h[:,1024:1536]
__global__ void k_att(const float* __restrict__ hdec, const float* __restrict__ pm,
                      const float* __restrict__ mem, const float* __restrict__ attq,
                      const float* __restrict__ attv,
                      _Float16* __restrict__ ri_h, _Float16* __restrict__ hc_h) {
  __shared__ float pq[cATT];
  __shared__ float sc[cS];
  __shared__ float red[256];
  int b = blockIdx.x, tid = threadIdx.x;
  if (tid < cATT) {
    float a = 0.0f;
    for (int k = 0; k < cDEC; ++k) a += attq[tid * cDEC + k] * hdec[b * cDEC + k];
    pq[tid] = a;
  }
  __syncthreads();
  {
    float sacc = 0.0f;
    const float* pmr = pm + ((size_t)(tid * cB + b)) * cATT;   // s = tid
    for (int a = 0; a < cATT; ++a) sacc += tanhf(pq[a] + pmr[a]) * attv[a];
    sc[tid] = sacc;
  }
  __syncthreads();
  red[tid] = sc[tid]; __syncthreads();
  for (int off = 128; off > 0; off >>= 1) {
    if (tid < off) red[tid] = fmaxf(red[tid], red[tid + off]);
    __syncthreads();
  }
  float mx = red[0];
  __syncthreads();
  float e = __expf(sc[tid] - mx);
  sc[tid] = e;
  red[tid] = e; __syncthreads();
  for (int off = 128; off > 0; off >>= 1) {
    if (tid < off) red[tid] += red[tid + off];
    __syncthreads();
  }
  float winv = 1.0f / red[0];
  for (int d = tid; d < cENC; d += 256) {
    float c = 0.0f;
    for (int s = 0; s < cS; ++s) c += sc[s] * mem[((size_t)(s * cB + b)) * cENC + d];
    c *= winv;
    ri_h[b * 768 + 256 + d]    = (_Float16)c;
    hc_h[b * 1536 + cDEC + d]  = (_Float16)c;
  }
}

// decoder LSTM cell elementwise; gates (32,4096) order i,f,g,o
__global__ void k_cell(const float* __restrict__ gates, float* __restrict__ cdec,
                       float* __restrict__ hdec, _Float16* __restrict__ hdec_h,
                       _Float16* __restrict__ hc_h) {
  int p = blockIdx.x * 256 + threadIdx.x;
  if (p >= cB * cDEC) return;
  int b = p >> 10, j = p & 1023;
  const float* g = gates + (size_t)b * 4 * cDEC;
  float c = sigm(g[cDEC + j]) * cdec[p] + sigm(g[j]) * tanhf(g[2 * cDEC + j]);
  float h = sigm(g[3 * cDEC + j]) * tanhf(c);
  cdec[p] = c; hdec[p] = h;
  hdec_h[p] = (_Float16)h;
  hc_h[b * 1536 + j] = (_Float16)h;
}

// scatter mo (32,240) of step t into mel_outputs (B,80,1500) + gate outputs
__global__ void k_mo_scatter(const float* __restrict__ mo, float* __restrict__ melout,
                             float* __restrict__ gateout, int t) {
  int p = blockIdx.x * 256 + threadIdx.x;
  if (p >= cB * 240) return;
  int b = p / 240, q = p % 240, r = q / cNMEL, m = q % cNMEL;
  float v = mo[p];
  melout[(size_t)b * cNMEL * cL + (size_t)m * cL + t * 3 + r] = v;
  if (m == cNMEL - 1) gateout[b * cL + t * 3 + r] = sigm(v);
}

// postnet conv1 im2col from mel_outputs; rows (b,t'), cols i*5+k (pad to 416)
__global__ void k_post_im2col(const float* __restrict__ melout, _Float16* __restrict__ X) {
  size_t idx = (size_t)blockIdx.x * 256 + threadIdx.x;
  if (idx >= (size_t)cB * cL * cKP) return;
  int row = (int)(idx / cKP), col = (int)(idx % cKP);
  float v = 0.0f;
  if (col < cNMEL * 5) {
    int b = row / cL, tp = row % cL;
    int i = col / 5, k = col % 5;
    int tt = tp + k - 2;
    if (tt >= 0 && tt < cL) v = melout[(size_t)b * cNMEL * cL + (size_t)i * cL + tt];
  }
  X[idx] = (_Float16)v;
}

// postnet conv2 (direct) + residual add into mel_postnet
__global__ void k_conv2(const float* __restrict__ y, const float* __restrict__ w,
                        const float* __restrict__ bias, const float* __restrict__ melout,
                        float* __restrict__ melpost) {
  int p = blockIdx.x * 256 + threadIdx.x;
  if (p >= cB * cNMEL * cL) return;
  int b = p / (cNMEL * cL), rem = p % (cNMEL * cL), o = rem / cL, t0 = rem % cL;
  float acc = bias[o];
  for (int k = 0; k < 5; ++k) {
    int tt = t0 + k - 2;
    if (tt < 0 || tt >= cL) continue;
    const float* yr = y + ((size_t)b * cL + tt) * 512;
    const float* wr = w + (size_t)o * 512 * 5 + k;
    for (int i = 0; i < 512; ++i) acc += wr[i * 5] * yr[i];
  }
  melpost[p] = acc + melout[p];
}

// ---------------------------------------------------------------------------
// host driver
// ---------------------------------------------------------------------------
static inline dim3 g1(size_t n) { return dim3((unsigned)((n + 255) / 256)); }

extern "C" void kernel_launch(void* const* d_in, const int* in_sizes, int n_in,
                              void* d_out, int out_size, void* d_ws, size_t ws_size,
                              hipStream_t stream) {
  (void)in_sizes; (void)n_in; (void)out_size; (void)ws_size;
  const int*   text   = (const int*)d_in[0];
  const float* mels   = (const float*)d_in[1];
  const float* emb    = (const float*)d_in[2];
  const float* conv1w = (const float*)d_in[3];
  const float* conv1b = (const float*)d_in[4];
  const float* bn1g   = (const float*)d_in[5];
  const float* bn1b   = (const float*)d_in[6];
  const float *lwih[6], *lwhh[6], *lbih[6], *lbhh[6];
  for (int i = 0; i < 6; ++i) {        // l-major, dir (f,b) minor
    lwih[i] = (const float*)d_in[7 + 4 * i];
    lwhh[i] = (const float*)d_in[8 + 4 * i];
    lbih[i] = (const float*)d_in[9 + 4 * i];
    lbhh[i] = (const float*)d_in[10 + 4 * i];
  }
  const float* prew1   = (const float*)d_in[31];
  const float* preb1   = (const float*)d_in[32];
  const float* prew2   = (const float*)d_in[33];
  const float* preb2   = (const float*)d_in[34];
  const float* attq    = (const float*)d_in[35];
  const float* attm    = (const float*)d_in[36];
  const float* attv    = (const float*)d_in[37];
  const float* cellwih = (const float*)d_in[38];
  const float* cellwhh = (const float*)d_in[39];
  const float* cellbih = (const float*)d_in[40];
  const float* cellbhh = (const float*)d_in[41];
  const float* outw    = (const float*)d_in[42];
  const float* outb    = (const float*)d_in[43];
  const float* post1w  = (const float*)d_in[44];
  const float* post1b  = (const float*)d_in[45];
  const float* pbng    = (const float*)d_in[46];
  const float* pbnb    = (const float*)d_in[47];
  const float* post2w  = (const float*)d_in[48];
  const float* post2b  = (const float*)d_in[49];

  float* melout  = (float*)d_out;                        // (B,80,1500)
  float* melpost = melout + (size_t)cB * cNMEL * cL;     // (B,80,1500)
  float* gateout = melpost + (size_t)cB * cNMEL * cL;    // (B,1500)

  char* ws = (char*)d_ws;
  size_t off = 0;
  auto take = [&](size_t bytes) -> char* {
    char* p = ws + off;
    off += (bytes + 255) & ~(size_t)255;
    return p;
  };
  // persistent f16 weights + small decoder state
  _Float16* conv1w_h = (_Float16*)take((size_t)cENC * cK1 * 2);
  _Float16 *wih_h[6], *whh_h[6]; float* cbias[6];
  for (int i = 0; i < 6; ++i) {
    wih_h[i] = (_Float16*)take((size_t)4 * cH * cENC * 2);
    whh_h[i] = (_Float16*)take((size_t)4 * cH * cH * 2);
    cbias[i] = (float*)take((size_t)4 * cH * 4);
  }
  _Float16* cellwih_h = (_Float16*)take((size_t)4 * cDEC * 768 * 2);
  _Float16* cellwhh_h = (_Float16*)take((size_t)4 * cDEC * cDEC * 2);
  float*    cellbias  = (float*)take((size_t)4 * cDEC * 4);
  _Float16* outw_h    = (_Float16*)take((size_t)240 * 1536 * 2);
  _Float16* post1w_h  = (_Float16*)take((size_t)512 * cKP * 2);
  _Float16* attm_h    = (_Float16*)take((size_t)cATT * cENC * 2);
  float* bns1 = (float*)take(512 * 4);
  float* bns2 = (float*)take(512 * 4);
  float*    hdec   = (float*)take((size_t)cB * cDEC * 4);
  float*    cdec   = (float*)take((size_t)cB * cDEC * 4);
  _Float16* hdec_h = (_Float16*)take((size_t)cB * cDEC * 2);
  _Float16* ri_h   = (_Float16*)take((size_t)cB * 768 * 2);
  _Float16* hc_h   = (_Float16*)take((size_t)cB * 1536 * 2);
  float*    gates  = (float*)take((size_t)cB * 4 * cDEC * 4);
  float*    mo     = (float*)take((size_t)cB * 240 * 4);
  // big region (encoder buffers; overlaid by postnet afterwards)
  size_t bigOff = off;
  _Float16* im2colE = (_Float16*)take((size_t)cB * cS * cK1 * 2);   // 41.9 MB
  float*    c1      = (float*)take((size_t)cB * cS * cENC * 4);     // 16.8 MB
  _Float16* xsA     = (_Float16*)take((size_t)cS * cB * cENC * 2);  //  8.4 MB
  _Float16* xsB     = (_Float16*)take((size_t)cS * cB * cENC * 2);  //  8.4 MB
  float*    xwF     = (float*)take((size_t)cS * cB * 4 * cH * 4);   // 33.6 MB
  float*    xwB     = (float*)take((size_t)cS * cB * 4 * cH * 4);   // 33.6 MB
  float*    memf    = (float*)take((size_t)cS * cB * cENC * 4);     // 16.8 MB
  float*    pmf     = (float*)take((size_t)cS * cB * cATT * 4);     //  4.2 MB
  // postnet overlays (encoder buffers dead by then)
  float*    ypost = (float*)(ws + bigOff);                 // 98.3 MB over im2colE..xwF
  _Float16* pim   = (_Float16*)xwB;                        // 39.9 MB over xwB+memf

  // ---- weight conversion to f16 ----
  k_cvt_f16<<<g1((size_t)cENC * cK1), 256, 0, stream>>>(conv1w, conv1w_h, cENC * cK1);
  for (int i = 0; i < 6; ++i) {
    k_cvt_f16<<<g1(4 * cH * cENC), 256, 0, stream>>>(lwih[i], wih_h[i], 4 * cH * cENC);
    k_cvt_f16<<<g1(4 * cH * cH), 256, 0, stream>>>(lwhh[i], whh_h[i], 4 * cH * cH);
    k_add_vec<<<g1(4 * cH), 256, 0, stream>>>(lbih[i], lbhh[i], cbias[i], 4 * cH);
  }
  k_cvt_f16<<<g1((size_t)4 * cDEC * 768), 256, 0, stream>>>(cellwih, cellwih_h, 4 * cDEC * 768);
  k_cvt_f16<<<g1((size_t)4 * cDEC * cDEC), 256, 0, stream>>>(cellwhh, cellwhh_h, 4 * cDEC * cDEC);
  k_add_vec<<<g1(4 * cDEC), 256, 0, stream>>>(cellbih, cellbhh, cellbias, 4 * cDEC);
  k_cvt_f16<<<g1(240 * 1536), 256, 0, stream>>>(outw, outw_h, 240 * 1536);
  k_cvt_pad<<<g1(512 * cKP), 256, 0, stream>>>(post1w, post1w_h, 512, 400, cKP);
  k_cvt_f16<<<g1(cATT * cENC), 256, 0, stream>>>(attm, attm_h, cATT * cENC);

  // ---- encoder: embedding -> conv(relu) -> BN -> 3x biLSTM ----
  k_emb_im2col<<<g1((size_t)cB * cS * cK1), 256, 0, stream>>>(text, emb, im2colE);
  k_gemm4<1, 1, 0><<<dim3(cENC / 64, cB * cS / 16), 32, 0, stream>>>(
      im2colE, conv1w_h, conv1b, c1, cB * cS, cENC, cK1);
  k_bn_stats<<<512, 256, 0, stream>>>(c1, cB * cS, bns1, bns2);
  k_bn_apply<<<g1((size_t)cB * cS * 512), 256, 0, stream>>>(c1, cB * cS, bns1, bns2,
                                                            bn1g, bn1b, 0);
  k_enc_transpose<<<g1((size_t)cB * cS * cENC), 256, 0, stream>>>(c1, xsA);

  const size_t lstmSmem = (size_t)cB * cH * 2 + (size_t)cB * cH * 4 + (size_t)cB * 4 * cH * 4;
  _Float16* xc = xsA;
  _Float16* xn = xsB;
  for (int l = 0; l < 3; ++l) {
    k_gemm4<0, 1, 0><<<dim3(4 * cH / 64, cS * cB / 16), 32, 0, stream>>>(
        xc, wih_h[2 * l + 0], cbias[2 * l + 0], xwF, cS * cB, 4 * cH, cENC);
    k_gemm4<0, 1, 0><<<dim3(4 * cH / 64, cS * cB / 16), 32, 0, stream>>>(
        xc, wih_h[2 * l + 1], cbias[2 * l + 1], xwB, cS * cB, 4 * cH, cENC);
    k_lstm<<<2, 1024, lstmSmem, stream>>>(xwF, xwB, whh_h[2 * l], whh_h[2 * l + 1],
                                          xn, memf);
    _Float16* t = xc; xc = xn; xn = t;
  }
  // processed memory pm = memory @ att_m^T   (rows = (s,b))
  k_gemm4<0, 0, 0><<<dim3(cATT / 64, cS * cB / 16), 32, 0, stream>>>(
      xc, attm_h, nullptr, pmf, cS * cB, cATT, cENC);

  // ---- decoder ----
  k_zero_f32<<<g1(cB * cDEC), 256, 0, stream>>>(hdec, cB * cDEC);
  k_zero_f32<<<g1(cB * cDEC), 256, 0, stream>>>(cdec, cB * cDEC);
  k_zero_f16<<<g1(cB * cDEC), 256, 0, stream>>>(hdec_h, cB * cDEC);
  for (int t = 0; t < cT; ++t) {
    k_prenet<<<cB, 256, 0, stream>>>(mels, t, prew1, preb1, prew2, preb2, ri_h);
    k_att<<<cB, 256, 0, stream>>>(hdec, pmf, memf, attq, attv, ri_h, hc_h);
    k_gemm4<0, 1, 0><<<dim3(4 * cDEC / 64, cB / 16), 32, 0, stream>>>(
        ri_h, cellwih_h, cellbias, gates, cB, 4 * cDEC, 768);
    k_gemm4<0, 0, 1><<<dim3(4 * cDEC / 64, cB / 16), 32, 0, stream>>>(
        hdec_h, cellwhh_h, nullptr, gates, cB, 4 * cDEC, cDEC);
    k_cell<<<g1(cB * cDEC), 256, 0, stream>>>(gates, cdec, hdec, hdec_h, hc_h);
    k_gemm<0, 1, 0><<<dim3(240 / 16, cB / 16), 32, 0, stream>>>(
        hc_h, outw_h, outb, mo, cB, 240, 1536);
    k_mo_scatter<<<g1(cB * 240), 256, 0, stream>>>(mo, melout, gateout, t);
  }

  // ---- postnet ----
  k_post_im2col<<<g1((size_t)cB * cL * cKP), 256, 0, stream>>>(melout, pim);
  k_gemm4<0, 1, 0><<<dim3(512 / 64, cB * cL / 16), 32, 0, stream>>>(
      pim, post1w_h, post1b, ypost, cB * cL, 512, cKP);
  k_bn_stats<<<512, 256, 0, stream>>>(ypost, cB * cL, bns1, bns2);
  k_bn_apply<<<g1((size_t)cB * cL * 512), 256, 0, stream>>>(ypost, cB * cL, bns1, bns2,
                                                            pbng, pbnb, 1);
  k_conv2<<<g1((size_t)cB * cNMEL * cL), 256, 0, stream>>>(ypost, post2w, post2b,
                                                           melout, melpost);
}
